// SymphoniesDecoder_51161650430131
// MI455X (gfx1250) — compile-verified, matrix operand order checked by reading, not run on hardware
//
#include <hip/hip_runtime.h>
#include <hip/hip_bf16.h>
#include <math.h>

#define NVOX   32768
#define NI     100
#define NTOK   10080
#define IMGH   384
#define IMGW   1280

typedef __attribute__((ext_vector_type(16))) _Float16 v16h;
typedef __attribute__((ext_vector_type(8)))  _Float16 v8h;
typedef __attribute__((ext_vector_type(8)))  float    v8f;

#define LSTR 40   // LDS row stride in halfs (80B: 16B-aligned, bank-spreading)

// ---------------------------------------------------------------------------
// Tiled WMMA GEMM: Y[M,N] = A[M,K] @ W[K,N] + bias (+res) (optional ReLU)
// Block: 256 threads = 8 waves; block tile 64x64; wave tile 16x32 (2 WMMAs
// sharing one A fragment); K stepped by 32 via LDS-staged f16 tiles.
// Requires K % 32 == 0 and N % 32 == 0 (true for all call sites); M ragged OK.
// ---------------------------------------------------------------------------
__global__ __launch_bounds__(256)
void gemm_wmma_kernel(const float* __restrict__ A, const float* __restrict__ W,
                      const float* __restrict__ bias, const float* __restrict__ res,
                      float* __restrict__ Y, int M, int K, int N, int relu)
{
    __shared__ _Float16 sA[64 * LSTR];   // sA[m][k]
    __shared__ _Float16 sB[64 * LSTR];   // sB[n][k]  (transposed weight tile)
    const int m0   = blockIdx.x * 64;
    const int n0   = blockIdx.y * 64;
    const int tid  = threadIdx.x;
    const int wave = tid >> 5;
    const int lane = tid & 31;
    const int wm   = (wave & 3) << 4;   // 4 wave-rows of 16
    const int wn   = (wave >> 2) << 5;  // 2 wave-cols of 32
    const int hl   = lane >> 4;         // lane half
    const int r    = lane & 15;

    // branch-free staging indices (clamped, never OOB; pad rows never stored)
    const int arow = tid >> 2;                    // 64 rows, 4 thr/row
    const int aseg = (tid & 3) << 3;              // 8 floats each
    const int am   = min(m0 + arow, M - 1);
    const int bkr  = tid >> 3;                    // 32 k-rows, 8 thr/row
    const int bseg = (tid & 7) << 3;              // 8 cols each
    const int bg   = min(n0 + bseg, N - 8);       // clamped global col base

    v8f acc0 = {0.f, 0.f, 0.f, 0.f, 0.f, 0.f, 0.f, 0.f};
    v8f acc1 = {0.f, 0.f, 0.f, 0.f, 0.f, 0.f, 0.f, 0.f};

    for (int kk = 0; kk < K; kk += 32) {
        // --- stage A tile: two float4 loads -> one 16B LDS store ---
        const float* ap = A + (long)am * K + kk + aseg;
        float4 a0 = *(const float4*)ap;
        float4 a1 = *(const float4*)(ap + 4);
        v8h av;
        av[0] = (_Float16)a0.x; av[1] = (_Float16)a0.y;
        av[2] = (_Float16)a0.z; av[3] = (_Float16)a0.w;
        av[4] = (_Float16)a1.x; av[5] = (_Float16)a1.y;
        av[6] = (_Float16)a1.z; av[7] = (_Float16)a1.w;
        *(v8h*)&sA[arow * LSTR + aseg] = av;

        // --- stage W tile transposed: two float4 loads -> 8 b16 stores ---
        const float* wp = W + (long)(kk + bkr) * N + bg;
        float4 w0 = *(const float4*)wp;
        float4 w1 = *(const float4*)(wp + 4);
        sB[(bseg + 0) * LSTR + bkr] = (_Float16)w0.x;
        sB[(bseg + 1) * LSTR + bkr] = (_Float16)w0.y;
        sB[(bseg + 2) * LSTR + bkr] = (_Float16)w0.z;
        sB[(bseg + 3) * LSTR + bkr] = (_Float16)w0.w;
        sB[(bseg + 4) * LSTR + bkr] = (_Float16)w1.x;
        sB[(bseg + 5) * LSTR + bkr] = (_Float16)w1.y;
        sB[(bseg + 6) * LSTR + bkr] = (_Float16)w1.z;
        sB[(bseg + 7) * LSTR + bkr] = (_Float16)w1.w;
        __syncthreads();

        // --- fragments: ISA A layout = two contiguous 8-half runs per lane ---
        // lane(hl,r): k in {hl*8..hl*8+7} U {hl*8+16..hl*8+23}
        v8h alo = *(const v8h*)&sA[(wm + r) * LSTR + hl * 8];
        v8h ahi = *(const v8h*)&sA[(wm + r) * LSTR + hl * 8 + 16];
        v16h af = __builtin_shufflevector(alo, ahi, 0, 1, 2, 3, 4, 5, 6, 7,
                                                    8, 9, 10, 11, 12, 13, 14, 15);
        v8h b0lo = *(const v8h*)&sB[(wn + r) * LSTR + hl * 8];
        v8h b0hi = *(const v8h*)&sB[(wn + r) * LSTR + hl * 8 + 16];
        v16h bf0 = __builtin_shufflevector(b0lo, b0hi, 0, 1, 2, 3, 4, 5, 6, 7,
                                                       8, 9, 10, 11, 12, 13, 14, 15);
        v8h b1lo = *(const v8h*)&sB[(wn + 16 + r) * LSTR + hl * 8];
        v8h b1hi = *(const v8h*)&sB[(wn + 16 + r) * LSTR + hl * 8 + 16];
        v16h bf1 = __builtin_shufflevector(b1lo, b1hi, 0, 1, 2, 3, 4, 5, 6, 7,
                                                       8, 9, 10, 11, 12, 13, 14, 15);
        acc0 = __builtin_amdgcn_wmma_f32_16x16x32_f16(false, af, false, bf0,
                                                      (short)0, acc0, false, false);
        acc1 = __builtin_amdgcn_wmma_f32_16x16x32_f16(false, af, false, bf1,
                                                      (short)0, acc1, false, false);
        __syncthreads();
    }

    // D layout: VGPR j -> row (hl*8 + j), col = lane%16
    int gn0 = n0 + wn + r;
    int gn1 = gn0 + 16;
    float bv0 = (gn0 < N && bias) ? bias[gn0] : 0.f;
    float bv1 = (gn1 < N && bias) ? bias[gn1] : 0.f;
    #pragma unroll
    for (int j = 0; j < 8; ++j) {
        int gm = m0 + wm + (hl << 3) + j;
        if (gm < M) {
            if (gn0 < N) {
                float v = acc0[j] + bv0;
                if (relu) v = fmaxf(v, 0.f);
                if (res)  v += res[(long)gm * N + gn0];
                Y[(long)gm * N + gn0] = v;
            }
            if (gn1 < N) {
                float v = acc1[j] + bv1;
                if (relu) v = fmaxf(v, 0.f);
                if (res)  v += res[(long)gm * N + gn1];
                Y[(long)gm * N + gn1] = v;
            }
        }
    }
}

// ---------------------------------------------------------------------------
// Softmax attention, Nk <= 128.  Block (32,8): one wave per (query, head).
// Branch-free: key index clamped (always-valid row), validity folded into
// score/weight selects -> straight-line b128 loads, no EXEC manipulation.
// ---------------------------------------------------------------------------
__global__ __launch_bounds__(256)
void attn_kernel(const float* __restrict__ Qp, const float* __restrict__ Kp,
                 const float* __restrict__ Vp, float* __restrict__ O,
                 int Nq, int Nk)
{
    int q = blockIdx.x;
    if (q >= Nq) return;
    int lane = threadIdx.x;
    int h    = threadIdx.y;

    const float4* qp = (const float4*)(Qp + (long)q * 128 + h * 16);
    float4 qv[4];
    #pragma unroll
    for (int i = 0; i < 4; ++i) qv[i] = qp[i];

    float sc[4];
    float mx = -1e30f;
    #pragma unroll
    for (int t = 0; t < 4; ++t) {
        int k  = lane + t * 32;
        int kc = min(k, Nk - 1);
        const float4* kp = (const float4*)(Kp + (long)kc * 128 + h * 16);
        float d = 0.f;
        #pragma unroll
        for (int i = 0; i < 4; ++i) {
            float4 kv = kp[i];
            d += qv[i].x * kv.x + qv[i].y * kv.y + qv[i].z * kv.z + qv[i].w * kv.w;
        }
        sc[t] = (k < Nk) ? d * 0.25f : -1e30f;   // 1/sqrt(16)
        mx = fmaxf(mx, sc[t]);
    }
    #pragma unroll
    for (int off = 16; off >= 1; off >>= 1) mx = fmaxf(mx, __shfl_xor(mx, off, 32));
    float sum = 0.f;
    #pragma unroll
    for (int t = 0; t < 4; ++t) {
        int k = lane + t * 32;
        sc[t] = (k < Nk) ? __expf(sc[t] - mx) : 0.f;
        sum += sc[t];
    }
    #pragma unroll
    for (int off = 16; off >= 1; off >>= 1) sum += __shfl_xor(sum, off, 32);
    float inv = 1.f / sum;

    float ov[16];
    #pragma unroll
    for (int c = 0; c < 16; ++c) ov[c] = 0.f;
    #pragma unroll
    for (int t = 0; t < 4; ++t) {
        int k  = lane + t * 32;
        int kc = min(k, Nk - 1);
        float w = sc[t] * inv;                    // 0 for invalid keys
        const float4* vp = (const float4*)(Vp + (long)kc * 128 + h * 16);
        #pragma unroll
        for (int i = 0; i < 4; ++i) {
            float4 vv = vp[i];
            ov[i * 4 + 0] += w * vv.x;
            ov[i * 4 + 1] += w * vv.y;
            ov[i * 4 + 2] += w * vv.z;
            ov[i * 4 + 3] += w * vv.w;
        }
    }
    #pragma unroll
    for (int c = 0; c < 16; ++c) {
        #pragma unroll
        for (int off = 16; off >= 1; off >>= 1) ov[c] += __shfl_xor(ov[c], off, 32);
    }
    if (lane == 0) {
        #pragma unroll
        for (int c = 0; c < 16; ++c) O[(long)q * 128 + h * 16 + c] = ov[c];
    }
}

// ---------------------------------------------------------------------------
// LayerNorm over 128 channels.  Block (32,8): one wave per row.
// ---------------------------------------------------------------------------
__global__ __launch_bounds__(256)
void ln_kernel(const float* __restrict__ X, const float* __restrict__ g,
               const float* __restrict__ b, float* __restrict__ Y, int M)
{
    int row = blockIdx.x * 8 + threadIdx.y;
    if (row >= M) return;
    int lane = threadIdx.x;
    float v[4];
    float s = 0.f;
    #pragma unroll
    for (int i = 0; i < 4; ++i) { v[i] = X[(long)row * 128 + lane * 4 + i]; s += v[i]; }
    #pragma unroll
    for (int off = 16; off >= 1; off >>= 1) s += __shfl_xor(s, off, 32);
    float mean = s * (1.f / 128.f);
    float vs = 0.f;
    #pragma unroll
    for (int i = 0; i < 4; ++i) { float d = v[i] - mean; vs += d * d; }
    #pragma unroll
    for (int off = 16; off >= 1; off >>= 1) vs += __shfl_xor(vs, off, 32);
    float rstd = rsqrtf(vs * (1.f / 128.f) + 1e-5f);
    #pragma unroll
    for (int i = 0; i < 4; ++i) {
        int c = lane * 4 + i;
        Y[(long)row * 128 + c] = (v[i] - mean) * rstd * g[c] + b[c];
    }
}

// ---------------------------------------------------------------------------
// Deformable 3D sampling (scene grid 64x64x8), H=8 L=1 P=8, fused aw-softmax.
// Block (16,8): thread = (channel, head); grid = Nq.
// ---------------------------------------------------------------------------
__global__ __launch_bounds__(128)
void msda3d_kernel(const float* __restrict__ val, const float* __restrict__ off,
                   const float* __restrict__ awl, const float* __restrict__ ref,
                   float* __restrict__ out, int Nq)
{
    int q = blockIdx.x;
    if (q >= Nq) return;
    int c = threadIdx.x, h = threadIdx.y;
    float w[8];
    float mx = -1e30f;
    #pragma unroll
    for (int p = 0; p < 8; ++p) { w[p] = awl[(long)q * 64 + h * 8 + p]; mx = fmaxf(mx, w[p]); }
    float s = 0.f;
    #pragma unroll
    for (int p = 0; p < 8; ++p) { w[p] = __expf(w[p] - mx); s += w[p]; }
    float inv = 1.f / s;
    float rx = ref[(long)q * 3 + 0], ry = ref[(long)q * 3 + 1], rz = ref[(long)q * 3 + 2];
    float acc = 0.f;
    #pragma unroll
    for (int p = 0; p < 8; ++p) {
        long oi = (long)q * 192 + (h * 8 + p) * 3;
        float x = (rx + off[oi + 0] * (1.f / 64.f)) * 64.f - 0.5f;
        float y = (ry + off[oi + 1] * (1.f / 64.f)) * 64.f - 0.5f;
        float z = (rz + off[oi + 2] * (1.f / 8.f)) * 8.f - 0.5f;
        float x0 = floorf(x), y0 = floorf(y), z0 = floorf(z);
        float wp = w[p] * inv;
        #pragma unroll
        for (int dx = 0; dx < 2; ++dx)
        #pragma unroll
        for (int dy = 0; dy < 2; ++dy)
        #pragma unroll
        for (int dz = 0; dz < 2; ++dz) {
            float xi = x0 + dx, yi = y0 + dy, zi = z0 + dz;
            float wt = (1.f - fabsf(x - xi)) * (1.f - fabsf(y - yi)) * (1.f - fabsf(z - zi));
            bool valid = (xi >= 0.f) & (xi < 64.f) & (yi >= 0.f) & (yi < 64.f) &
                         (zi >= 0.f) & (zi < 8.f);
            int xc = min(max((int)xi, 0), 63);
            int yc = min(max((int)yi, 0), 63);
            int zc = min(max((int)zi, 0), 7);
            long idx = ((long)xc * 64 + yc) * 8 + zc;
            acc += wp * (valid ? wt : 0.f) * val[idx * 128 + h * 16 + c];
        }
    }
    out[(long)q * 128 + h * 16 + c] = acc;
}

// ---------------------------------------------------------------------------
// Deformable 2D sampling over 3 pyramid levels, H=8 L=3 P=4, fused softmax(12).
// ---------------------------------------------------------------------------
__global__ __launch_bounds__(128)
void msda2d_kernel(const float* __restrict__ val, const float* __restrict__ off,
                   const float* __restrict__ awl, const float* __restrict__ ref,
                   float* __restrict__ out, int Nq)
{
    const int Hs[3] = {48, 24, 12};
    const int Ws[3] = {160, 80, 40};
    const int st[3] = {0, 7680, 9600};
    int q = blockIdx.x;
    if (q >= Nq) return;
    int c = threadIdx.x, h = threadIdx.y;
    float w[12];
    float mx = -1e30f;
    #pragma unroll
    for (int j = 0; j < 12; ++j) { w[j] = awl[(long)q * 96 + h * 12 + j]; mx = fmaxf(mx, w[j]); }
    float s = 0.f;
    #pragma unroll
    for (int j = 0; j < 12; ++j) { w[j] = __expf(w[j] - mx); s += w[j]; }
    float inv = 1.f / s;
    float rx = ref[(long)q * 2 + 0], ry = ref[(long)q * 2 + 1];
    float acc = 0.f;
    #pragma unroll
    for (int l = 0; l < 3; ++l) {
        float nw = (float)Ws[l], nh = (float)Hs[l];
        #pragma unroll
        for (int p = 0; p < 4; ++p) {
            long oi = (long)q * 192 + ((h * 3 + l) * 4 + p) * 2;
            float x = (rx + off[oi + 0] / nw) * nw - 0.5f;
            float y = (ry + off[oi + 1] / nh) * nh - 0.5f;
            float x0 = floorf(x), y0 = floorf(y);
            float wp = w[l * 4 + p] * inv;
            #pragma unroll
            for (int dy = 0; dy < 2; ++dy)
            #pragma unroll
            for (int dx = 0; dx < 2; ++dx) {
                float xi = x0 + dx, yi = y0 + dy;
                float wt = (1.f - fabsf(x - xi)) * (1.f - fabsf(y - yi));
                bool valid = (xi >= 0.f) & (xi < nw) & (yi >= 0.f) & (yi < nh);
                int xc = min(max((int)xi, 0), Ws[l] - 1);
                int yc = min(max((int)yi, 0), Hs[l] - 1);
                long idx = st[l] + (long)yc * Ws[l] + xc;
                acc += wp * (valid ? wt : 0.f) * val[idx * 128 + h * 16 + c];
            }
        }
    }
    out[(long)q * 128 + h * 16 + c] = acc;
}

// ---------------------------------------------------------------------------
// Geometry helpers + misc kernels
// ---------------------------------------------------------------------------
__device__ inline void inv3x3(const float* K, float* I)
{
    float a = K[0], b = K[1], c = K[2], d = K[3], e = K[4], f = K[5],
          g = K[6], h = K[7], i = K[8];
    float A = e * i - f * h, B = -(d * i - f * g), C = d * h - e * g;
    float inv = 1.f / (a * A + b * B + c * C);
    I[0] = A * inv;            I[1] = -(b * i - c * h) * inv; I[2] = (b * f - c * e) * inv;
    I[3] = B * inv;            I[4] = (a * i - c * g) * inv;  I[5] = -(a * f - c * d) * inv;
    I[6] = C * inv;            I[7] = -(a * h - b * g) * inv; I[8] = (a * e - b * d) * inv;
}

__device__ inline void pix2vox(float px, float py, float d, const float* Ki,
                               const float* Em, const float* vo, float offset,
                               int* vx, int* vy, int* vz)
{
    float X = px * d, Yc = py * d, Z = d;
    float pcx = Ki[0] * X + Ki[1] * Yc + Ki[2] * Z;
    float pcy = Ki[3] * X + Ki[4] * Yc + Ki[5] * Z;
    float pcz = Ki[6] * X + Ki[7] * Yc + Ki[8] * Z;
    // E is rigid [R|t]: inv(E)*[pc,1] = R^T (pc - t)
    float qx = pcx - Em[3], qy = pcy - Em[7], qz = pcz - Em[11];
    float pwx = Em[0] * qx + Em[4] * qy + Em[8] * qz;
    float pwy = Em[1] * qx + Em[5] * qy + Em[9] * qz;
    float pwz = Em[2] * qx + Em[6] * qy + Em[10] * qz;
    *vx = (int)((pwx - vo[0]) * 5.f - offset);
    *vy = (int)((pwy - vo[1]) * 5.f - offset);
    *vz = (int)((pwz - vo[2]) * 5.f - offset);
}

__global__ void pack_feats_kernel(const float* __restrict__ f0, const float* __restrict__ f1,
                                  const float* __restrict__ f2, float* __restrict__ out)
{
    int i = blockIdx.x * 256 + threadIdx.x;
    if (i >= NTOK * 128) return;
    int t = i >> 7, ch = i & 127;
    const float* f; int n, start;
    if (t < 7680)      { f = f0; n = 7680; start = 0; }
    else if (t < 9600) { f = f1; n = 1920; start = 7680; }
    else               { f = f2; n = 480;  start = 9600; }
    out[i] = f[(long)ch * n + (t - start)];
}

__global__ void refpix_kernel(const int* __restrict__ pp, float* __restrict__ out)
{
    int i = blockIdx.x * 256 + threadIdx.x;
    if (i >= NVOX) return;
    out[i * 2 + 0] = ((float)pp[i * 2 + 1] + 0.5f) * (1.f / (float)IMGW);
    out[i * 2 + 1] = ((float)pp[i * 2 + 0] + 0.5f) * (1.f / (float)IMGH);
}

__global__ void refvox_kernel(float* __restrict__ out)
{
    int i = blockIdx.x * 256 + threadIdx.x;
    if (i >= NVOX) return;
    int gx = i >> 9, gy = (i >> 3) & 63, gz = i & 7;
    out[i * 3 + 0] = ((float)gx + 0.5f) * (1.f / 64.f);
    out[i * 3 + 1] = ((float)gy + 0.5f) * (1.f / 64.f);
    out[i * 3 + 2] = ((float)gz + 0.5f) * (1.f / 8.f);
}

__global__ void ref3d_kernel(const float* __restrict__ rp, const float* __restrict__ depth,
                             const float* __restrict__ Km, const float* __restrict__ Em,
                             const float* __restrict__ vo, float* __restrict__ out)
{
    int i = threadIdx.x;
    if (i >= NI) return;
    float Ki[9];
    inv3x3(Km, Ki);
    float px = rp[i * 2 + 0] * (float)IMGW;
    float py = rp[i * 2 + 1] * (float)IMGH;
    int cx = min(max((int)px, 0), IMGW - 1);
    int cy = min(max((int)py, 0), IMGH - 1);
    float d = depth[cy * IMGW + cx];
    int vx, vy, vz;
    pix2vox(px, py, d, Ki, Em, vo, 0.f, &vx, &vy, &vz);
    out[i * 3 + 0] = (float)vx * (1.f / 64.f);
    out[i * 3 + 1] = (float)vy * (1.f / 64.f);
    out[i * 3 + 2] = (float)vz * (1.f / 8.f);
}

__global__ void ptsmask_kernel(const float* __restrict__ depth, const float* __restrict__ Km,
                               const float* __restrict__ Em, const float* __restrict__ vo,
                               int* __restrict__ mask)
{
    int i = blockIdx.x * 256 + threadIdx.x;
    if (i >= IMGH * IMGW) return;
    int r = i / IMGW, c = i % IMGW;
    float Ki[9];
    inv3x3(Km, Ki);
    int vx, vy, vz;
    pix2vox((float)c, (float)r, depth[i], Ki, Em, vo, 0.5f, &vx, &vy, &vz);
    if (vx >= 0 && vx < 64 && vy >= 0 && vy < 64 && vz >= 0 && vz < 8)
        atomicOr(&mask[(vx * 64 + vy) * 8 + vz], 1);
}

__global__ void copy_kernel(const float* __restrict__ src, float* __restrict__ dst, long n)
{
    long i = (long)blockIdx.x * 256 + threadIdx.x;
    if (i < n) dst[i] = src[i];
}

__global__ void zero_int_kernel(int* __restrict__ p, int n)
{
    int i = blockIdx.x * 256 + threadIdx.x;
    if (i < n) p[i] = 0;
}

__global__ void select_int_kernel(const int* __restrict__ mask, const float* __restrict__ A,
                                  const float* __restrict__ B, float* __restrict__ Y, int rows)
{
    long i = (long)blockIdx.x * 256 + threadIdx.x;
    if (i >= (long)rows * 128) return;
    Y[i] = mask[i >> 7] ? A[i] : B[i];
}

__global__ void select_u8_kernel(const unsigned char* __restrict__ mask,
                                 const float* __restrict__ A, const float* __restrict__ B,
                                 float* __restrict__ Y, int rows)
{
    long i = (long)blockIdx.x * 256 + threadIdx.x;
    if (i >= (long)rows * 128) return;
    Y[i] = mask[i >> 7] ? A[i] : B[i];
}

__global__ __launch_bounds__(128)
void cls_kernel(const float* __restrict__ scene, const float* __restrict__ W,
                const float* __restrict__ b, float* __restrict__ out)
{
    int v = blockIdx.x * 128 + threadIdx.x;
    if (v >= NVOX) return;
    const float* row = scene + (long)v * 128;
    #pragma unroll 4
    for (int k = 0; k < 20; ++k) {
        float s = b[k];
        #pragma unroll 8
        for (int c = 0; c < 128; ++c) s += row[c] * W[c * 20 + k];
        out[(long)k * NVOX + v] = s;
    }
}

// ---------------------------------------------------------------------------
// Host orchestration
// ---------------------------------------------------------------------------
struct DLP { const float *W1,*W2,*Wa,*Woff,*Wout,*Wv,*b1,*b1n,*b2,*b2n,*ba,*boff,*bout,*bv,*g1,*g2; };
struct TLP { const float *W1,*W2,*Wk,*Wo,*Wq,*Wv,*b1,*b1n,*b2,*b2n,*bk,*bo,*bq,*bv,*g1,*g2; };

extern "C" void kernel_launch(void* const* d_in, const int* in_sizes, int n_in,
                              void* d_out, int out_size, void* d_ws, size_t ws_size,
                              hipStream_t stream)
{
    (void)in_sizes; (void)n_in; (void)out_size; (void)ws_size;

    const float* inst_in  = (const float*)d_in[0];
    const float* ref2d    = (const float*)d_in[1];
    const float* ref3dpix = (const float*)d_in[2];
    const float* feat0    = (const float*)d_in[3];
    const float* feat1    = (const float*)d_in[4];
    const float* feat2    = (const float*)d_in[5];
    const float* depth    = (const float*)d_in[6];
    const float* Km       = (const float*)d_in[7];
    const float* Em       = (const float*)d_in[8];
    const float* vo       = (const float*)d_in[9];
    const int*   ppix     = (const int*)d_in[10];
    const unsigned char* fov = (const unsigned char*)d_in[11];

    // params pytree flattened: alphabetical dict keys, lists in order
    int pi = 12;
    auto nx = [&]() -> const float* { return (const float*)d_in[pi++]; };
    const float* cls_W = nx();
    const float* cls_b = nx();
    auto getDL = [&]() { DLP p; p.W1=nx(); p.W2=nx(); p.Wa=nx(); p.Woff=nx(); p.Wout=nx();
                         p.Wv=nx(); p.b1=nx(); p.b1n=nx(); p.b2=nx(); p.b2n=nx(); p.ba=nx();
                         p.boff=nx(); p.bout=nx(); p.bv=nx(); p.g1=nx(); p.g2=nx(); return p; };
    auto getTL = [&]() { TLP p; p.W1=nx(); p.W2=nx(); p.Wk=nx(); p.Wo=nx(); p.Wq=nx();
                         p.Wv=nx(); p.b1=nx(); p.b1n=nx(); p.b2=nx(); p.b2n=nx(); p.bk=nx();
                         p.bo=nx(); p.bq=nx(); p.bv=nx(); p.g1=nx(); p.g2=nx(); return p; };
    DLP Lqida[3], Lqsda[3], Lssda[3];
    TLP Lqsa[3],  Lsqca[3];
    for (int i = 0; i < 3; ++i) {
        Lqida[i] = getDL();
        Lqsa[i]  = getTL();
        Lqsda[i] = getDL();
        Lsqca[i] = getTL();
        Lssda[i] = getDL();
    }
    DLP prop = getDL();
    const float* scene_embed = nx();

    // workspace carving (floats)
    float* wsf = (float*)d_ws;
    size_t wo = 0;
    auto alloc = [&](size_t n) { float* p = wsf + wo; wo += n; return p; };
    float* featflat = alloc((size_t)NTOK * 128);
    float* refpix   = alloc((size_t)NVOX * 2);
    float* refvox   = alloc((size_t)NVOX * 3);
    float* ref3db   = alloc(512);
    int*   ptsmask  = (int*)alloc(NVOX);
    float* scene    = alloc((size_t)NVOX * 128);
    float* sqb      = alloc((size_t)NVOX * 128);
    float* vbuf     = alloc((size_t)NVOX * 128);
    float* msout    = alloc((size_t)NVOX * 128);
    float* tmp1     = alloc((size_t)NVOX * 128);
    float* tmp2     = alloc((size_t)NVOX * 128);
    float* hid      = alloc((size_t)NVOX * 512);
    float* offb     = alloc((size_t)NVOX * 192);
    float* awb      = alloc((size_t)NVOX * 96);
    float* instb    = alloc(128 * 128);
    float* iQ       = alloc(128 * 128);
    float* iK       = alloc(128 * 128);
    float* iV       = alloc(128 * 128);
    float* ims      = alloc(128 * 128);
    float* it1      = alloc(128 * 128);
    float* it2      = alloc(128 * 128);
    float* ihid     = alloc(128 * 512);
    float* ioff     = alloc(128 * 192);
    float* iaw      = alloc(128 * 96);

    auto gemm = [&](const float* A, const float* W, const float* b, const float* res,
                    float* Y, int M, int K, int N, int relu) {
        dim3 g((M + 63) / 64, (N + 63) / 64);
        gemm_wmma_kernel<<<g, 256, 0, stream>>>(A, W, b, res, Y, M, K, N, relu);
    };
    auto lnorm = [&](const float* X, const float* g, const float* b, float* Y, int M) {
        ln_kernel<<<dim3((M + 7) / 8), dim3(32, 8), 0, stream>>>(X, g, b, Y, M);
    };
    // shared deformable/transformer tail: out-proj(+res) -> LN -> FFN(+res) -> LN
    auto dl_tail = [&](const float* Wout, const float* bout, const float* g1, const float* b1n,
                       const float* W1, const float* b1, const float* W2, const float* b2,
                       const float* g2, const float* b2n, const float* qin, const float* ms,
                       float* t1, float* t2, float* hb, float* outb, int M) {
        gemm(ms, Wout, bout, qin, t2, M, 128, 128, 0);
        lnorm(t2, g1, b1n, t1, M);
        gemm(t1, W1, b1, nullptr, hb, M, 128, 512, 1);
        gemm(hb, W2, b2, t1, t2, M, 512, 128, 0);
        lnorm(t2, g2, b2n, outb, M);
    };

    // ---- preprocessing ----
    pack_feats_kernel<<<(NTOK * 128 + 255) / 256, 256, 0, stream>>>(feat0, feat1, feat2, featflat);
    refpix_kernel<<<(NVOX + 255) / 256, 256, 0, stream>>>(ppix, refpix);
    refvox_kernel<<<(NVOX + 255) / 256, 256, 0, stream>>>(refvox);
    ref3d_kernel<<<1, 128, 0, stream>>>(ref3dpix, depth, Km, Em, vo, ref3db);
    zero_int_kernel<<<(NVOX + 255) / 256, 256, 0, stream>>>(ptsmask, NVOX);
    ptsmask_kernel<<<(IMGH * IMGW + 255) / 256, 256, 0, stream>>>(depth, Km, Em, vo, ptsmask);
    copy_kernel<<<(NVOX * 128 + 255) / 256, 256, 0, stream>>>(scene_embed, scene, (long)NVOX * 128);
    copy_kernel<<<(NI * 128 + 255) / 256, 256, 0, stream>>>(inst_in, instb, (long)NI * 128);

    // ---- proposal deformable layer (scene queries over image features) ----
    gemm(featflat, prop.Wv, prop.bv, nullptr, vbuf, NTOK, 128, 128, 0);
    gemm(scene, prop.Woff, prop.boff, nullptr, offb, NVOX, 128, 192, 0);
    gemm(scene, prop.Wa, prop.ba, nullptr, awb, NVOX, 128, 96, 0);
    msda2d_kernel<<<NVOX, dim3(16, 8), 0, stream>>>(vbuf, offb, awb, refpix, msout, NVOX);
    dl_tail(prop.Wout, prop.bout, prop.g1, prop.b1n, prop.W1, prop.b1, prop.W2, prop.b2,
            prop.g2, prop.b2n, scene, msout, tmp1, tmp2, hid, sqb, NVOX);
    select_int_kernel<<<(NVOX * 128 + 255) / 256, 256, 0, stream>>>(ptsmask, sqb, scene, scene, NVOX);

    // ---- decoder layers ----
    for (int i = 0; i < 3; ++i) {
        // qsa: instance self-attention
        gemm(instb, Lqsa[i].Wq, Lqsa[i].bq, nullptr, iQ, NI, 128, 128, 0);
        gemm(instb, Lqsa[i].Wk, Lqsa[i].bk, nullptr, iK, NI, 128, 128, 0);
        gemm(instb, Lqsa[i].Wv, Lqsa[i].bv, nullptr, iV, NI, 128, 128, 0);
        attn_kernel<<<NI, dim3(32, 8), 0, stream>>>(iQ, iK, iV, ims, NI, NI);
        dl_tail(Lqsa[i].Wo, Lqsa[i].bo, Lqsa[i].g1, Lqsa[i].b1n, Lqsa[i].W1, Lqsa[i].b1,
                Lqsa[i].W2, Lqsa[i].b2, Lqsa[i].g2, Lqsa[i].b2n,
                instb, ims, it1, it2, ihid, instb, NI);

        // sqca: scene cross-attention over instances
        gemm(scene, Lsqca[i].Wq, Lsqca[i].bq, nullptr, tmp1, NVOX, 128, 128, 0);
        gemm(instb, Lsqca[i].Wk, Lsqca[i].bk, nullptr, iK, NI, 128, 128, 0);
        gemm(instb, Lsqca[i].Wv, Lsqca[i].bv, nullptr, iV, NI, 128, 128, 0);
        attn_kernel<<<NVOX, dim3(32, 8), 0, stream>>>(tmp1, iK, iV, msout, NVOX, NI);
        dl_tail(Lsqca[i].Wo, Lsqca[i].bo, Lsqca[i].g1, Lsqca[i].b1n, Lsqca[i].W1, Lsqca[i].b1,
                Lsqca[i].W2, Lsqca[i].b2, Lsqca[i].g2, Lsqca[i].b2n,
                scene, msout, tmp1, tmp2, hid, sqb, NVOX);

        // ssda: scene self deformable 3D attention (value = pre-update scene)
        gemm(scene, Lssda[i].Wv, Lssda[i].bv, nullptr, vbuf, NVOX, 128, 128, 0);
        gemm(sqb, Lssda[i].Woff, Lssda[i].boff, nullptr, offb, NVOX, 128, 192, 0);
        gemm(sqb, Lssda[i].Wa, Lssda[i].ba, nullptr, awb, NVOX, 128, 64, 0);
        msda3d_kernel<<<NVOX, dim3(16, 8), 0, stream>>>(vbuf, offb, awb, refvox, msout, NVOX);
        dl_tail(Lssda[i].Wout, Lssda[i].bout, Lssda[i].g1, Lssda[i].b1n, Lssda[i].W1, Lssda[i].b1,
                Lssda[i].W2, Lssda[i].b2, Lssda[i].g2, Lssda[i].b2n,
                sqb, msout, tmp1, tmp2, hid, sqb, NVOX);
        select_u8_kernel<<<(NVOX * 128 + 255) / 256, 256, 0, stream>>>(fov, sqb, scene, scene, NVOX);

        // qsda: instance deformable 3D attention over scene
        gemm(scene, Lqsda[i].Wv, Lqsda[i].bv, nullptr, vbuf, NVOX, 128, 128, 0);
        gemm(instb, Lqsda[i].Woff, Lqsda[i].boff, nullptr, ioff, NI, 128, 192, 0);
        gemm(instb, Lqsda[i].Wa, Lqsda[i].ba, nullptr, iaw, NI, 128, 64, 0);
        msda3d_kernel<<<NI, dim3(16, 8), 0, stream>>>(vbuf, ioff, iaw, ref3db, ims, NI);
        dl_tail(Lqsda[i].Wout, Lqsda[i].bout, Lqsda[i].g1, Lqsda[i].b1n, Lqsda[i].W1, Lqsda[i].b1,
                Lqsda[i].W2, Lqsda[i].b2, Lqsda[i].g2, Lqsda[i].b2n,
                instb, ims, it1, it2, ihid, instb, NI);

        // qida: instance deformable 2D attention over image features
        gemm(featflat, Lqida[i].Wv, Lqida[i].bv, nullptr, vbuf, NTOK, 128, 128, 0);
        gemm(instb, Lqida[i].Woff, Lqida[i].boff, nullptr, ioff, NI, 128, 192, 0);
        gemm(instb, Lqida[i].Wa, Lqida[i].ba, nullptr, iaw, NI, 128, 96, 0);
        msda2d_kernel<<<NI, dim3(16, 8), 0, stream>>>(vbuf, ioff, iaw, ref2d, ims, NI);
        dl_tail(Lqida[i].Wout, Lqida[i].bout, Lqida[i].g1, Lqida[i].b1n, Lqida[i].W1, Lqida[i].b1,
                Lqida[i].W2, Lqida[i].b2, Lqida[i].g2, Lqida[i].b2n,
                instb, ims, it1, it2, ihid, instb, NI);
    }

    // ---- classifier head: [NVOX,128] @ [128,20] -> out[k][vox] ----
    cls_kernel<<<(NVOX + 127) / 128, 128, 0, stream>>>(scene, cls_W, cls_b, (float*)d_out);
}
